// GCN_16896401342680
// MI455X (gfx1250) — matrix operand order repaired and checked by reading, not compiled, not run on hardware
//
#include <hip/hip_runtime.h>

typedef _Float16 v16h __attribute__((ext_vector_type(16)));
typedef _Float16 v8h  __attribute__((ext_vector_type(8)));
typedef float    v8f  __attribute__((ext_vector_type(8)));
typedef unsigned int uv4 __attribute__((ext_vector_type(4)));
typedef int          iv8 __attribute__((ext_vector_type(8)));
typedef int          iv4 __attribute__((ext_vector_type(4)));

#define N_NODES   100000
#define N_EDGES   1600000
#define IN_C      256
#define HID       128
#define NEG_SLOPE 0.01f

// ---------------- degree / normalization ----------------

__global__ void k_fill_ones(float* __restrict__ deg, int n) {
    int i = blockIdx.x * blockDim.x + threadIdx.x;
    if (i < n) deg[i] = 1.0f;   // self-loop contributes 1 to every degree
}

__global__ void k_deg(const int* __restrict__ dst, float* __restrict__ deg, int e) {
    int i = blockIdx.x * blockDim.x + threadIdx.x;
    if (i < e) atomicAdd(&deg[dst[i]], 1.0f);
}

__global__ void k_rsqrt(float* __restrict__ d, int n) {
    int i = blockIdx.x * blockDim.x + threadIdx.x;
    if (i < n) d[i] = rsqrtf(d[i]);   // deg >= 1 always (self loops)
}

__global__ void k_norm(const int* __restrict__ src, const int* __restrict__ dst,
                       const float* __restrict__ dinv, float* __restrict__ norm, int e) {
    int i = blockIdx.x * blockDim.x + threadIdx.x;
    if (i < e) norm[i] = dinv[src[i]] * dinv[dst[i]];
}

// ---------------- fp32 -> fp16 activation conversion ----------------

__global__ void k_to_f16(const float* __restrict__ in, _Float16* __restrict__ out, int total) {
    int idx = blockIdx.x * blockDim.x + threadIdx.x;
    if (idx < total) out[idx] = (_Float16)in[idx];
}

// ---------------- pack W into WMMA B-fragment layout ----------------
// Fragment f = kt*8+nt. Lane l (half=l/16, lr=l%16) holds col n=nt*16+lr,
// K = kt*32 + half*16 + i  (i=0..15), stored as 16 contiguous halves per lane.
__global__ void k_pack_w(const float* __restrict__ W, _Float16* __restrict__ wp, int fin) {
    int idx = blockIdx.x * blockDim.x + threadIdx.x;
    if (idx < fin * HID) {
        int f    = idx >> 9;        // fragment (512 halves each)
        int rem  = idx & 511;
        int lane = rem >> 4;
        int i    = rem & 15;
        int kt   = f >> 3, nt = f & 7;
        int half = lane >> 4, lr = lane & 15;
        int k    = kt * 32 + half * 16 + i;
        int n    = nt * 16 + lr;
        wp[idx] = (_Float16)W[k * HID + n];
    }
}

// ---------------- dense transform: hw = h @ W (WMMA f16 -> f32) ----------------
// Block = 256 threads = 8 waves; wave = 16-row strip x 128 cols (8 tiles),
// K stepped by 32. Packed f16 weights staged LDS via Tensor Data Mover.

union ABFrag { v16h v; v8h h[2]; };

__global__ void k_gemm_wmma(const _Float16* __restrict__ h16,
                            const _Float16* __restrict__ wp,
                            float* __restrict__ hw, int fin, int nRowTiles) {
    extern __shared__ _Float16 wlds[];   // packed fragments, fin*HID halves
    const int tid = threadIdx.x;
    const int totalBytes = fin * HID * 2;

#if __has_builtin(__builtin_amdgcn_tensor_load_to_lds)
    if (tid < 32) {
        // ---- Tensor DMA: contiguous 1D tile of 8B elements, global -> LDS 0 ----
        unsigned long long ga = (unsigned long long)(size_t)wp;
        unsigned ntd = (unsigned)(totalBytes >> 3);   // elements of 8 bytes
        uv4 g0;
        g0[0] = 1u;                                    // count=1 valid descriptor
        g0[1] = 0u;                                    // lds_addr = 0 (dynamic LDS base)
        g0[2] = (unsigned)(ga & 0xFFFFFFFFull);        // global_addr[31:0]
        g0[3] = (unsigned)((ga >> 32) & 0x1FFFFFFull)  // global_addr[56:32]
              | (2u << 30);                            // type = 2 (image)
        iv8 g1;
        g1[0] = (int)(3u << 16);                       // data_size=3 (8B), mask=0
        g1[1] = (int)((ntd & 0xFFFFu) << 16);          // tensor_dim0[15:0]
        g1[2] = (int)(((ntd >> 16) & 0xFFFFu) | (1u << 16)); // dim0[31:16] | tensor_dim1=1
        g1[3] = (int)((ntd & 0xFFFFu) << 16);          // tile_dim0
        g1[4] = (int)(1u);                             // tile_dim1=1, tile_dim2=0
        g1[5] = (int)ntd;                              // tensor_dim0_stride[31:0]
        g1[6] = 0;                                     // stride0 hi | stride1 lo
        g1[7] = 0;
        iv4 z4 = {0, 0, 0, 0};
        iv8 z8 = {0, 0, 0, 0, 0, 0, 0, 0};
        // amdgpu-toolchain (clang-23) 6-arg form:
        // (uint32x4 g0, int32x8 g1, int32x4, int32x4, int32x8, i32 cpol)
        __builtin_amdgcn_tensor_load_to_lds(g0, g1, z4, z4, z8, 0);
#if __has_builtin(__builtin_amdgcn_s_wait_tensorcnt)
        __builtin_amdgcn_s_wait_tensorcnt(0);
#else
        asm volatile("s_wait_tensorcnt 0x0" ::: "memory");
#endif
    }
#else
    for (int idx = tid; idx < (totalBytes >> 4); idx += blockDim.x)
        ((uint4*)wlds)[idx] = ((const uint4*)wp)[idx];
#endif
    __syncthreads();

    const int wave = tid >> 5;
    const int lane = tid & 31;
    const int half = lane >> 4;
    const int lr   = lane & 15;

    const int rowTile = blockIdx.x * 8 + wave;
    if (rowTile < nRowTiles) {
        v8f zero = {0.f, 0.f, 0.f, 0.f, 0.f, 0.f, 0.f, 0.f};
        v8f acc[8];
        #pragma unroll
        for (int nt = 0; nt < 8; ++nt) acc[nt] = zero;

        const _Float16* __restrict__ hrow = h16 + (size_t)(rowTile * 16 + lr) * fin;
        const _Float16* __restrict__ bl   = wlds + lane * 16;
        const int kTiles = fin >> 5;

        for (int kt = 0; kt < kTiles; ++kt) {
            // A fragment: lane holds row lr, K = half*16 + 0..15
            const _Float16* p = hrow + kt * 32 + half * 16;
            ABFrag a;
            a.h[0] = *(const v8h*)(p);
            a.h[1] = *(const v8h*)(p + 8);

            const _Float16* bk = bl + kt * 8 * 512;   // fragment stride = 512 halves
            #pragma unroll
            for (int nt = 0; nt < 8; ++nt) {
                ABFrag b;
                b.h[0] = *(const v8h*)(bk + nt * 512);
                b.h[1] = *(const v8h*)(bk + nt * 512 + 8);
                acc[nt] = __builtin_amdgcn_wmma_f32_16x16x32_f16(
                    false, a.v, false, b.v, (short)0, acc[nt], false, false);
            }
        }

        const int rowBase = rowTile * 16;
        #pragma unroll
        for (int nt = 0; nt < 8; ++nt) {
            #pragma unroll
            for (int r = 0; r < 8; ++r) {
                hw[(size_t)(rowBase + half * 8 + r) * HID + nt * 16 + lr] = acc[nt][r];
            }
        }
    }
}

// ---------------- aggregation ----------------

// agg[i][f] = bias[f] + hw[i][f] * dinv[i]^2   (self-loop term)
__global__ void k_init_agg(const float* __restrict__ hw, const float* __restrict__ bias,
                           const float* __restrict__ dinv, float* __restrict__ agg, int n) {
    int idx = blockIdx.x * blockDim.x + threadIdx.x;
    if (idx < n * HID) {
        int i = idx >> 7;
        int f = idx & 127;
        float di = dinv[i];
        agg[idx] = bias[f] + hw[idx] * di * di;
    }
}

// 8 edges per 256-thread block; each thread handles 4 features of one edge.
__global__ void k_scatter(const int* __restrict__ src, const int* __restrict__ dst,
                          const float* __restrict__ norm, const float* __restrict__ hw,
                          float* __restrict__ agg, int e) {
    const int t  = threadIdx.x;
    const int f4 = t & 31;
    const int eo = t >> 5;
    const int ed = blockIdx.x * 8 + eo;
    if (ed < e) {
        const int   s = src[ed];
        const int   d = dst[ed];
        const float w = norm[ed];
        const float4 v = *(const float4*)(hw + (size_t)s * HID + f4 * 4);
        float* out = agg + (size_t)d * HID + f4 * 4;
        atomicAdd(out + 0, v.x * w);
        atomicAdd(out + 1, v.y * w);
        atomicAdd(out + 2, v.z * w);
        atomicAdd(out + 3, v.w * w);
    }
}

__global__ void k_leaky16(const float* __restrict__ agg, _Float16* __restrict__ out, int total) {
    int idx = blockIdx.x * blockDim.x + threadIdx.x;
    if (idx < total) {
        float v = agg[idx];
        out[idx] = (_Float16)(v > 0.f ? v : NEG_SLOPE * v);
    }
}

__global__ void k_leaky32(const float* __restrict__ agg, float* __restrict__ out, int total) {
    int idx = blockIdx.x * blockDim.x + threadIdx.x;
    if (idx < total) {
        float v = agg[idx];
        out[idx] = v > 0.f ? v : NEG_SLOPE * v;
    }
}

// ---------------- driver ----------------

extern "C" void kernel_launch(void* const* d_in, const int* in_sizes, int n_in,
                              void* d_out, int out_size, void* d_ws, size_t ws_size,
                              hipStream_t stream) {
    const float* x   = (const float*)d_in[0];
    const int*   ei  = (const int*)d_in[1];
    const int*   src = ei;
    const int*   dst = ei + N_EDGES;
    const float* W1 = (const float*)d_in[2]; const float* b1 = (const float*)d_in[3];
    const float* W2 = (const float*)d_in[4]; const float* b2 = (const float*)d_in[5];
    const float* W3 = (const float*)d_in[6]; const float* b3 = (const float*)d_in[7];
    const float* W4 = (const float*)d_in[8]; const float* b4 = (const float*)d_in[9];
    float* out = (float*)d_out;

    float* ws   = (float*)d_ws;
    float* dinv = ws;                                   // N
    float* norm = dinv + N_NODES;                       // E
    float* hw   = norm + N_EDGES;                       // N*HID
    float* agg  = hw   + (size_t)N_NODES * HID;         // N*HID
    _Float16* x16   = (_Float16*)(agg + (size_t)N_NODES * HID);  // N*IN_C
    _Float16* act16 = x16 + (size_t)N_NODES * IN_C;              // N*HID
    _Float16* wp    = act16 + (size_t)N_NODES * HID;             // IN_C*HID (packed W)

    // --- GCN normalization (once) ---
    k_fill_ones<<<(N_NODES + 255) / 256, 256, 0, stream>>>(dinv, N_NODES);
    k_deg      <<<(N_EDGES + 255) / 256, 256, 0, stream>>>(dst, dinv, N_EDGES);
    k_rsqrt    <<<(N_NODES + 255) / 256, 256, 0, stream>>>(dinv, N_NODES);
    k_norm     <<<(N_EDGES + 255) / 256, 256, 0, stream>>>(src, dst, dinv, norm, N_EDGES);

    // --- input to f16 once ---
    k_to_f16<<<(N_NODES * IN_C + 255) / 256, 256, 0, stream>>>(x, x16, N_NODES * IN_C);

    const int nRowTiles = N_NODES / 16;                 // 6250 (exact)
    const int gemmGrid  = (nRowTiles + 7) / 8;
    const int elemGrid  = (N_NODES * HID + 255) / 256;
    const int scatGrid  = (N_EDGES + 7) / 8;

    auto layer = [&](const _Float16* hin16, int fin, const float* W, const float* b,
                     bool last) {
        k_pack_w<<<(fin * HID + 255) / 256, 256, 0, stream>>>(W, wp, fin);
        size_t ldsBytes = (size_t)fin * HID * 2;
        k_gemm_wmma<<<gemmGrid, 256, ldsBytes, stream>>>(hin16, wp, hw, fin, nRowTiles);
        k_init_agg <<<elemGrid, 256, 0, stream>>>(hw, b, dinv, agg, N_NODES);
        k_scatter  <<<scatGrid, 256, 0, stream>>>(src, dst, norm, hw, agg, N_EDGES);
        if (last) k_leaky32<<<elemGrid, 256, 0, stream>>>(agg, out, N_NODES * HID);
        else      k_leaky16<<<elemGrid, 256, 0, stream>>>(agg, act16, N_NODES * HID);
    };

    layer(x16,   IN_C, W1, b1, false);
    layer(act16, HID,  W2, b2, false);
    layer(act16, HID,  W3, b3, false);
    layer(act16, HID,  W4, b4, true);
}
